// AttentionSentGRU_25847113187759
// MI455X (gfx1250) — matrix operand order repaired and checked by reading, not compiled, not run on hardware
//
#include <hip/hip_runtime.h>
#include <hip/hip_bf16.h>

// ---------------- problem constants ----------------
constexpr int BB = 128;   // batch
constexpr int TT = 1024;  // seq len
constexpr int DD = 256;   // input dim
constexpr int HH = 128;   // hidden
constexpr int G3 = 384;   // 3*H (gates r,z,n)
constexpr int E2 = 256;   // 2*H
constexpr int CC = 10;    // classes

// ---------------- WMMA types ----------------
typedef __attribute__((ext_vector_type(16))) __bf16 bf16x16;
typedef __attribute__((ext_vector_type(8)))  float  f32x8;
typedef __attribute__((__vector_size__(16))) int    vint4;  // async builtin ptr type

union ABfrag { bf16x16 v; unsigned int u[8]; };
union Cfrag  { f32x8 v; float f[8]; };

__device__ __forceinline__ unsigned short f2bf(float x) {
  unsigned u = __float_as_uint(x);
  return (unsigned short)((u + 0x7FFFu + ((u >> 16) & 1u)) >> 16);  // RNE
}
__device__ __forceinline__ float bf2f(unsigned short h) {
  return __uint_as_float(((unsigned)h) << 16);
}

__device__ __forceinline__ f32x8 wmma_bf16(bf16x16 a, bf16x16 b, f32x8 c) {
  // v_wmma_f32_16x16x32_bf16 : (neg_a, A, neg_b, B, c_mod, C, reuse_a, reuse_b)
  return __builtin_amdgcn_wmma_f32_16x16x32_bf16(false, a, false, b, (short)0, c,
                                                 false, false);
}

// ---------------- CDNA5 async global<->LDS (ASYNCcnt) ----------------
#if __has_builtin(__builtin_amdgcn_global_load_async_to_lds_b128) && \
    __has_builtin(__builtin_amdgcn_global_store_async_from_lds_b128) && \
    __has_builtin(__builtin_amdgcn_s_wait_asynccnt)
#define USE_ASYNC 1
#else
#define USE_ASYNC 0
#endif

__device__ __forceinline__ void g2l_b128(void* lds_dst, const void* gsrc) {
#if USE_ASYNC
  __builtin_amdgcn_global_load_async_to_lds_b128((vint4*)gsrc, (vint4*)lds_dst, 0, 0);
#else
  *(uint4*)lds_dst = *(const uint4*)gsrc;
#endif
}
__device__ __forceinline__ void l2g_b128(void* gdst, const void* lds_src) {
#if USE_ASYNC
  __builtin_amdgcn_global_store_async_from_lds_b128((vint4*)gdst, (vint4*)lds_src, 0, 0);
#else
  *(uint4*)gdst = *(const uint4*)lds_src;
#endif
}
__device__ __forceinline__ void wait_async0() {
#if USE_ASYNC
  __builtin_amdgcn_s_wait_asynccnt(0);
#endif
}

// A fragment: 16x32 bf16 from row-major bf16 LDS.
// lanes 0-15: M=lane, K = kc*32 + {0..7, 16..23}; lanes 16-31: M=lane-16, +8.
__device__ __forceinline__ bf16x16 loadA_bf(const unsigned short* base, int lane,
                                            int rowStride, int kc) {
  ABfrag r;
  const int m  = lane & 15;
  const int kb = kc * 32 + ((lane >> 4) ? 8 : 0);
  const unsigned short* p = base + m * rowStride + kb;
#pragma unroll
  for (int v = 0; v < 4; ++v) r.u[v]     = *(const unsigned*)(p + 2 * v);
#pragma unroll
  for (int v = 0; v < 4; ++v) r.u[4 + v] = *(const unsigned*)(p + 16 + 2 * v);
  return r.v;
}

// B fragment: 32x16 bf16, weights stored native [n][k] (K-pair = one b32 load).
// lanes 0-15: N=nb+lane, K = kc*32 + {0..7, 16..23} ; lanes 16-31: K += 8.
__device__ __forceinline__ bf16x16 loadB_bf(const unsigned short* base, int lane,
                                            int rowStride, int kc, int nb) {
  ABfrag r;
  const int n  = nb + (lane & 15);
  const int kb = kc * 32 + ((lane >> 4) ? 8 : 0);
  const unsigned short* p = base + n * rowStride + kb;
#pragma unroll
  for (int v = 0; v < 4; ++v) r.u[v]     = *(const unsigned*)(p + 2 * v);
#pragma unroll
  for (int v = 0; v < 4; ++v) r.u[4 + v] = *(const unsigned*)(p + 16 + 2 * v);
  return r.v;
}

__device__ __forceinline__ float sigmoidf(float x) { return 1.f / (1.f + expf(-x)); }

// =====================================================================
// Kernel 1: fused bidirectional GRU scan.
// grid = (8 batch-chunks, 2 directions), block = 256 (8 waves).
// LDS: w_ih bf16 192K + w_hh bf16 96K + x stage 8K + h bf16 4K = 300 KB.
// w_hh B-fragments and biases resident in registers across all 1024 steps;
// exact fp32 h kept in registers (one owner lane per element); bf16 shadow
// in LDS feeds the WMMA A-fragments and the async out-store.
// =====================================================================
__global__ __launch_bounds__(256)
void gru_scan_kernel(const float* __restrict__ x,
                     const float* __restrict__ wih_f, const float* __restrict__ whh_f,
                     const float* __restrict__ bih_f, const float* __restrict__ bhh_f,
                     const float* __restrict__ wih_b, const float* __restrict__ whh_b,
                     const float* __restrict__ bih_b, const float* __restrict__ bhh_b,
                     unsigned short* __restrict__ outbf) {
  extern __shared__ char smem[];
  unsigned short* wih  = (unsigned short*)smem;     // [384][256] bf16
  unsigned short* whh  = wih + G3 * DD;             // [384][128] bf16
  unsigned short* xst  = whh + G3 * HH;             // [16][256]  bf16
  unsigned short* hbuf = xst + 16 * DD;             // [16][128]  bf16

  const int dir = blockIdx.y;
  const int m0  = blockIdx.x * 16;
  const int tid = threadIdx.x;

  const float* gwih = dir ? wih_b : wih_f;
  const float* gwhh = dir ? whh_b : whh_f;
  const float* gbih = dir ? bih_b : bih_f;
  const float* gbhh = dir ? bhh_b : bhh_f;

  for (int i = tid; i < G3 * DD; i += 256) wih[i] = f2bf(gwih[i]);
  for (int i = tid; i < G3 * HH; i += 256) whh[i] = f2bf(gwhh[i]);
  for (int i = tid; i < 16 * HH; i += 256) hbuf[i] = 0;
  __syncthreads();

  const int lane = tid & 31;
  const int wave = tid >> 5;
  const int col0 = wave * 16;          // 8 waves cover j = 0..127
  const int n    = lane & 15;
  const int j    = col0 + n;
  const int mh   = (lane >> 4) * 8;

  // per-lane gate biases (fixed j for the whole scan)
  const float bR  = gbih[j] + gbhh[j];
  const float bZ  = gbih[128 + j] + gbhh[128 + j];
  const float bNx = gbih[256 + j];
  const float bNh = gbhh[256 + j];

  // resident w_hh B-fragments: 3 gates x 4 k-chunks (96 VGPRs)
  bf16x16 wB[12];
#pragma unroll
  for (int g = 0; g < 3; ++g)
#pragma unroll
    for (int kc = 0; kc < 4; ++kc)
      wB[g * 4 + kc] = loadB_bf(whh, lane, HH, kc, col0 + 128 * g);

  float hreg[8];
#pragma unroll
  for (int rr = 0; rr < 8; ++rr) hreg[rr] = 0.f;

  const int srow = tid >> 4;           // staging: row 0..15
  const int scol = (tid & 15) * 16;    // 16 elems each

  for (int s = 0; s < TT; ++s) {
    const int t = dir ? (TT - 1 - s) : s;

    // stage x[m0..m0+15][t][:] -> bf16 LDS; prefetch next step's tile
    {
      const float* src = x + ((size_t)(m0 + srow) * TT + t) * DD + scol;
      unsigned short* dst = xst + srow * DD + scol;
#pragma unroll
      for (int i = 0; i < 16; ++i) dst[i] = f2bf(src[i]);
      if (s + 1 < TT) {
        const int tn = dir ? (TT - 2 - s) : (s + 1);
        __builtin_prefetch(x + ((size_t)(m0 + srow) * TT + tn) * DD + scol, 0, 0);
      }
    }
    __syncthreads();  // xst ready; previous step's hbuf writes visible

    Cfrag aR, aZ, aN, aHN;
    aR.v = f32x8{}; aZ.v = f32x8{}; aN.v = f32x8{}; aHN.v = f32x8{};

    // xg = x_t @ w_ih.T   (K = 256, 8 chunks)
#pragma unroll
    for (int kc = 0; kc < 8; ++kc) {
      bf16x16 a = loadA_bf(xst, lane, DD, kc);
      aR.v = wmma_bf16(a, loadB_bf(wih, lane, DD, kc, col0),       aR.v);
      aZ.v = wmma_bf16(a, loadB_bf(wih, lane, DD, kc, col0 + 128), aZ.v);
      aN.v = wmma_bf16(a, loadB_bf(wih, lane, DD, kc, col0 + 256), aN.v);
    }
    // hg = h @ w_hh.T     (K = 128, 4 chunks; resident B frags)
#pragma unroll
    for (int kc = 0; kc < 4; ++kc) {
      bf16x16 a = loadA_bf(hbuf, lane, HH, kc);
      aR.v  = wmma_bf16(a, wB[kc],     aR.v);
      aZ.v  = wmma_bf16(a, wB[4 + kc], aZ.v);
      aHN.v = wmma_bf16(a, wB[8 + kc], aHN.v);
    }

    wait_async0();    // prior step's async out-store no longer reads hbuf
    __syncthreads();  // all hbuf/xst reads of this step complete

    // gates + state update (fp32 recurrence in registers, bf16 shadow in LDS)
#pragma unroll
    for (int rr = 0; rr < 8; ++rr) {
      float rg = sigmoidf(aR.f[rr] + bR);
      float zg = sigmoidf(aZ.f[rr] + bZ);
      float ng = tanhf(aN.f[rr] + bNx + rg * (aHN.f[rr] + bNh));
      hreg[rr] = (1.f - zg) * ng + zg * hreg[rr];
      hbuf[(mh + rr) * HH + j] = f2bf(hreg[rr]);
    }
    __syncthreads();  // full rows in hbuf before async store

    // async store this step's 16x128 bf16 output rows: 16B per thread
    {
      const int row = tid >> 4, ch = tid & 15;
      unsigned short* gdst =
          outbf + (((size_t)(m0 + row)) * TT + t) * E2 + dir * HH + ch * 8;
      l2g_b128(gdst, hbuf + row * HH + ch * 8);
    }
  }
  wait_async0();
}

// =====================================================================
// Kernel 2: attention scores  s[b,t] = sum_e tanh((out@W)[b,t,e]+bias[e])*q[e]
// W_sent transposed into LDS bf16 once; its B-fragments resident in registers;
// bf16 `out` tiles staged via async global->LDS; ds_add_f32 reduction.
// =====================================================================
__global__ __launch_bounds__(256)
void attn_score_kernel(const unsigned short* __restrict__ outbf,
                       const float* __restrict__ Wsent,
                       const float* __restrict__ bias, const float* __restrict__ query,
                       float* __restrict__ scores) {
  extern __shared__ char smem[];
  unsigned short* wt  = (unsigned short*)smem;      // [e][d] = Wsent[d][e], bf16
  unsigned short* ost = wt + E2 * E2;               // [16][256] bf16
  float* red = (float*)(ost + 16 * E2);             // [16]

  const int tid = threadIdx.x;
  for (int i = tid; i < E2 * E2; i += 256) {
    int d = i >> 8, e = i & 255;                    // Wsent row-major [d][e]
    wt[e * E2 + d] = f2bf(Wsent[i]);
  }
  __syncthreads();

  const int lane = tid & 31;
  const int wave = tid >> 5;
  const int n    = lane & 15;
  const int mh   = (lane >> 4) * 8;
  const int e0   = wave * 32 + n, e1 = e0 + 16;
  const float bb0 = bias[e0],  bb1 = bias[e1];
  const float qq0 = query[e0], qq1 = query[e1];

  // resident W_sent B-fragments: 2 n-tiles x 8 k-chunks (128 VGPRs)
  bf16x16 w0[8], w1[8];
#pragma unroll
  for (int kc = 0; kc < 8; ++kc) {
    w0[kc] = loadB_bf(wt, lane, E2, kc, wave * 32);
    w1[kc] = loadB_bf(wt, lane, E2, kc, wave * 32 + 16);
  }

  const int srow = tid >> 4;
  const int scol = (tid & 15) * 16;
  const int tiles = (BB * TT) / 16;                 // 8192 row-tiles

  for (int mt = blockIdx.x; mt < tiles; mt += gridDim.x) {
    {
      const unsigned short* src = outbf + ((size_t)mt * 16 + srow) * E2 + scol;
      unsigned short* dst = ost + srow * E2 + scol;
      g2l_b128(dst, src);
      g2l_b128(dst + 8, src + 8);
    }
    if (tid < 16) red[tid] = 0.f;
    wait_async0();
    __syncthreads();

    Cfrag a0, a1; a0.v = f32x8{}; a1.v = f32x8{};
#pragma unroll
    for (int kc = 0; kc < 8; ++kc) {
      bf16x16 a = loadA_bf(ost, lane, E2, kc);
      a0.v = wmma_bf16(a, w0[kc], a0.v);
      a1.v = wmma_bf16(a, w1[kc], a1.v);
    }
#pragma unroll
    for (int rr = 0; rr < 8; ++rr) {
      float v0 = tanhf(a0.f[rr] + bb0) * qq0;
      float v1 = tanhf(a1.f[rr] + bb1) * qq1;
      atomicAdd(&red[mh + rr], v0 + v1);            // ds_add_f32
    }
    __syncthreads();
    if (tid < 16) scores[(size_t)mt * 16 + tid] = red[tid];
    __syncthreads();
  }
}

// =====================================================================
// Kernel 3: softmax over T per batch row.
// =====================================================================
__global__ __launch_bounds__(256)
void softmax_kernel(const float* __restrict__ scores, float* __restrict__ attnw) {
  __shared__ float sred[256];
  const int b = blockIdx.x, tid = threadIdx.x;
  const float* s = scores + (size_t)b * TT;

  float mx = -3.4e38f;
  for (int t = tid; t < TT; t += 256) mx = fmaxf(mx, s[t]);
  sred[tid] = mx; __syncthreads();
  for (int off = 128; off; off >>= 1) {
    if (tid < off) sred[tid] = fmaxf(sred[tid], sred[tid + off]);
    __syncthreads();
  }
  mx = sred[0]; __syncthreads();

  float sum = 0.f;
  for (int t = tid; t < TT; t += 256) sum += expf(s[t] - mx);
  sred[tid] = sum; __syncthreads();
  for (int off = 128; off; off >>= 1) {
    if (tid < off) sred[tid] += sred[tid + off];
    __syncthreads();
  }
  const float inv = 1.f / sred[0];
  for (int t = tid; t < TT; t += 256) attnw[(size_t)b * TT + t] = expf(s[t] - mx) * inv;
}

// =====================================================================
// Kernel 4: pooled[b,:] = sum_t out[b,t,:]*attn[b,t]; then tiny final linear.
// =====================================================================
__global__ __launch_bounds__(256)
void pool_linear_kernel(const unsigned short* __restrict__ outbf,
                        const float* __restrict__ attnw,
                        const float* __restrict__ lin_w, const float* __restrict__ lin_b,
                        float* __restrict__ y) {
  __shared__ float pl[E2];
  const int b = blockIdx.x, d = threadIdx.x;
  const unsigned short* ob = outbf + (size_t)b * TT * E2 + d;
  const float* aw = attnw + (size_t)b * TT;
  float acc = 0.f;
  for (int t = 0; t < TT; ++t) acc += bf2f(ob[(size_t)t * E2]) * aw[t];
  pl[d] = acc;
  __syncthreads();
  if (d < CC) {
    float v = lin_b[d];
#pragma unroll 4
    for (int k = 0; k < E2; ++k) v += pl[k] * lin_w[d * E2 + k];
    y[b * CC + d] = v;
  }
}

// =====================================================================
extern "C" void kernel_launch(void* const* d_in, const int* in_sizes, int n_in,
                              void* d_out, int out_size, void* d_ws, size_t ws_size,
                              hipStream_t stream) {
  (void)in_sizes; (void)n_in; (void)out_size; (void)ws_size;
  const float* x      = (const float*)d_in[0];
  const float* wih_f  = (const float*)d_in[1];
  const float* whh_f  = (const float*)d_in[2];
  const float* bih_f  = (const float*)d_in[3];
  const float* bhh_f  = (const float*)d_in[4];
  const float* wih_b  = (const float*)d_in[5];
  const float* whh_b  = (const float*)d_in[6];
  const float* bih_b  = (const float*)d_in[7];
  const float* bhh_b  = (const float*)d_in[8];
  const float* Wsent  = (const float*)d_in[9];
  const float* bias_s = (const float*)d_in[10];
  const float* query  = (const float*)d_in[11];
  const float* lin_w  = (const float*)d_in[12];
  const float* lin_b  = (const float*)d_in[13];
  float* y = (float*)d_out;

  unsigned short* outbf = (unsigned short*)d_ws;           // [B,T,256] bf16 = 64 MB
  float* scores = (float*)(outbf + (size_t)BB * TT * E2);  // [B*T] f32
  float* attnw  = scores + (size_t)BB * TT;                // [B*T] f32

  const size_t lds_scan = (size_t)(G3 * DD + G3 * HH + 16 * DD + 16 * HH) * 2; // 300 KB
  const size_t lds_attn = (size_t)(E2 * E2 + 16 * E2) * 2 + 16 * 4;            // ~136 KB

  static bool attr_done = []() {
    (void)hipFuncSetAttribute((const void*)gru_scan_kernel,
                              hipFuncAttributeMaxDynamicSharedMemorySize, 310 * 1024);
    (void)hipFuncSetAttribute((const void*)attn_score_kernel,
                              hipFuncAttributeMaxDynamicSharedMemorySize, 160 * 1024);
    return true;
  }();
  (void)attr_done;

  gru_scan_kernel<<<dim3(BB / 16, 2), dim3(256), lds_scan, stream>>>(
      x, wih_f, whh_f, bih_f, bhh_f, wih_b, whh_b, bih_b, bhh_b, outbf);
  attn_score_kernel<<<dim3(512), dim3(256), lds_attn, stream>>>(
      outbf, Wsent, bias_s, query, scores);
  softmax_kernel<<<dim3(BB), dim3(256), 0, stream>>>(scores, attnw);
  pool_linear_kernel<<<dim3(BB), dim3(256), 0, stream>>>(outbf, attnw, lin_w, lin_b, y);
}
